// ECCONetwork_75144747810844
// MI455X (gfx1250) — matrix operand order repaired and checked by reading, not compiled, not run on hardware
//
#include <hip/hip_runtime.h>

typedef __attribute__((ext_vector_type(16))) __bf16 v16bf;
typedef __attribute__((ext_vector_type(8)))  __bf16 v8bf;
typedef __attribute__((ext_vector_type(8)))  float  v8f;

#define B_ 2
#define N_ 768
#define M_ 512
#define DT_ 0.1f
#define ANG8_ 0.78539816339744830962f   // 2*pi/8

// ---- workspace offsets (bytes) ----
#define OFF_P1   0LL          // (B,N,2) f32              12288
#define OFF_BOX  16384LL      // box feats bf16 1024x16   32768
#define OFF_Y    65536LL      // GEMM out f32 1536x192    1179648
#define OFF_OUT0 1310720LL    // activation buf f32       1179648
#define OFF_OUT1 2621440LL    // activation buf f32       1179648
#define OFF_AO   3932160LL    // obstacle A bf16 1536x800 2457600
#define OFF_AF   6553600LL    // fluid A bf16 1536x9408   28901376
#define OFF_B0o  35651584LL   // 800x64  bf16
#define OFF_B0f  35753984LL   // 2368x128 bf16
#define OFF_B1   36360192LL   // 9408x128 bf16
#define OFF_B2   38768640LL   // 6272x128 bf16
#define OFF_B3   40374272LL   // 6272x128 bf16
#define OFF_B4   41979904LL   // 6272x16  bf16

// ---------------------------------------------------------------------------
// Precompute B-operand matrices (Kfull + folded dense paths), bf16, in the
// per-lane B layout (contiguous k, as k_gemm's A): addr = ((k/32)*nct + col/16)*512 + lane*16 + slot,
// lane = (col%16) + 16*((k%32)>=16), slot = (k%32)%16.
// ---------------------------------------------------------------------------
__global__ __launch_bounds__(256)
void k_prepB(int layer, int total, int Kg, int cs8, int ncols, int Cc,
             const float* __restrict__ convW, const float* __restrict__ denseW,
             const float* __restrict__ extraW, __bf16* __restrict__ Bout)
{
    int idx = blockIdx.x * 256 + threadIdx.x;
    if (idx >= total) return;
    int k   = idx / ncols;
    int col = idx - k * ncols;
    float val = 0.f;

    if (layer <= 1) {
        // layer0 obstacle (0) / fluid (1): k_rho1_to_reg + (fluid) dense fold
        if (k < Kg) {
            int n = k / cs8, cs = k - (k / cs8) * cs8;
            if (cs < Cc * 2 && col < 64) {
                int c = cs >> 1, aa = cs & 1;
                int o = col >> 3, m = col & 7;
                int r = n >> 4, t = n & 15;
                int tt = (t - 2 * m + 16) & 15;
                const float* wp = &convW[(((o * Cc + c) * 3 + r) * 16 + tt) * 2];
                float sm, cm; __sincosf(ANG8_ * (float)m, &sm, &cm);
                val = (aa == 0) ? (wp[0] * cm - wp[1] * sm) : (wp[0] * sm + wp[1] * cm);
            }
        } else if (layer == 1) {
            int cs = k - Kg;
            if (cs < 38 && col >= 64) {
                int c = cs >> 1, aa = cs & 1;
                int o = (col - 64) >> 3, m = col & 7;
                float accv = 0.f;
                for (int s = 0; s < 8; ++s) {
                    float sm, cm; __sincosf(ANG8_ * (float)s, &sm, &cm);
                    float bas = (aa == 0) ? (extraW[0] * cm - extraW[1] * sm)
                                          : (extraW[0] * sm + extraW[1] * cm);
                    accv += bas * denseW[(o * Cc + c) * 8 + ((m - s + 8) & 7)];
                }
                val = accv;
            }
        }
    } else if (layer <= 4) {
        // conv1/2/3: k_reg_to_reg + dense fold
        int o = col >> 3, m = col & 7;
        if (k < Kg) {
            int n = k / cs8, cs = k - (k / cs8) * cs8;
            int c = cs >> 3, s = cs & 7;
            int r = n >> 4, t = n & 15;
            int tt = (t - 2 * m + 16) & 15;
            int ss = (s - m + 8) & 7;
            val = convW[(((o * Cc + c) * 3 + r) * 16 + tt) * 8 + ss];
        } else {
            int cs = k - Kg;
            int c = cs >> 3, s = cs & 7;
            val = denseW[(o * Cc + c) * 8 + ((m - s + 8) & 7)];
        }
    } else {
        // conv4: k_reg_to_rho1 + dense4 (bias term cancels: sum of basis == 0)
        if (col < 2) {
            int p = col;
            if (k < Kg) {
                int n = k / cs8, cs = k - (k / cs8) * cs8;
                int c = cs >> 3, s = cs & 7;
                int r = n >> 4, t = n & 15;
                float accv = 0.f;
                for (int m = 0; m < 8; ++m) {
                    int tt = (t - 2 * m + 16) & 15;
                    int ss = (s - m + 8) & 7;
                    const float* wp = &convW[(((c * 3 + r) * 16 + tt) * 8 + ss) * 2];
                    float sm, cm; __sincosf(ANG8_ * (float)m, &sm, &cm);
                    accv += (p == 0) ? (wp[0] * cm - wp[1] * sm) : (wp[0] * sm + wp[1] * cm);
                }
                val = accv;
            } else {
                int cs = k - Kg;
                int c = cs >> 3, s = cs & 7;
                float accv = 0.f;
                for (int m = 0; m < 8; ++m) {
                    float sm, cm; __sincosf(ANG8_ * (float)m, &sm, &cm);
                    float bas = (p == 0) ? (extraW[0] * cm - extraW[1] * sm)
                                         : (extraW[0] * sm + extraW[1] * cm);
                    accv += denseW[c * 8 + ((m - s + 8) & 7)] * bas;
                }
                val = accv;
            }
        }
    }

    int kc = k >> 5, kl = k & 31;
    int ct = col >> 4, cl = col & 15;
    int ln = cl + ((kl & 16) ? 16 : 0);
    int sl = kl & 15;
    long long off = ((long long)kc * (ncols >> 4) + ct) * 512 + ln * 16 + sl;
    Bout[off] = (__bf16)val;
}

// ---------------------------------------------------------------------------
// Integrate + build layer0 feature rows (bf16 ext regions) + state output.
// ---------------------------------------------------------------------------
__global__ __launch_bounds__(256)
void k_integrate(const float* __restrict__ p0, const float* __restrict__ v0,
                 const float* __restrict__ acc, const float* __restrict__ other,
                 const float* __restrict__ v0enc, const float* __restrict__ boxfeats,
                 float* __restrict__ p1, __bf16* __restrict__ afext,
                 __bf16* __restrict__ aoext, __bf16* __restrict__ boxbf,
                 float* __restrict__ stateOut)
{
    int bi = blockIdx.x * blockDim.x + threadIdx.x;
    if (bi < B_ * N_) {
        float v0x = v0[bi * 2 + 0], v0y = v0[bi * 2 + 1];
        float v1x = v0x + DT_ * acc[bi * 2 + 0];
        float v1y = v0y + DT_ * acc[bi * 2 + 1];
        p1[bi * 2 + 0] = p0[bi * 2 + 0] + DT_ * 0.5f * (v0x + v1x);
        p1[bi * 2 + 1] = p0[bi * 2 + 1] + DT_ * 0.5f * (v0y + v1y);
        __bf16* e = afext + (long long)bi * 2368;
        e[0] = (__bf16)v1x; e[1] = (__bf16)v1y;
        for (int c = 0; c < 4; ++c) e[2 + c] = (__bf16)other[bi * 4 + c];
        for (int t = 0; t < 32; ++t) {
            float v = v0enc[bi * 32 + t];
            e[6 + t] = (__bf16)v;
            stateOut[bi * 32 + t] = v;
        }
        for (int c = 38; c < 64; ++c) e[c] = (__bf16)0.f;
        __bf16* e2 = aoext + (long long)bi * 800;
        for (int c = 0; c < 32; ++c) e2[c] = (__bf16)0.f;
    }
    if (bi < B_ * M_) {
        __bf16* bb = boxbf + bi * 16;
        bb[0] = (__bf16)boxfeats[bi * 2 + 0];
        bb[1] = (__bf16)boxfeats[bi * 2 + 1];
        for (int c = 2; c < 16; ++c) bb[c] = (__bf16)0.f;
    }
}

// ---------------------------------------------------------------------------
// Neighbor aggregation via WMMA: G[bin][cs] = sum_j w(i,j,bin)*feats[j][cs].
// Block = 1 dest particle, 8 waves.  Bin-weight A-tiles (48x32) and feature
// B-tiles (32 x 16*nct) are built in LDS in native WMMA operand layouts.
// ---------------------------------------------------------------------------
__global__ __launch_bounds__(256)
void k_aggregate(const float* __restrict__ p1, const float* __restrict__ psrc,
                 const float* __restrict__ msrc, const __bf16* __restrict__ feats,
                 int fstride, int Nsrc, int nct,
                 __bf16* __restrict__ Aout, long long astride)
{
    __shared__ __align__(32) __bf16 Aw[3 * 512];
    __shared__ __align__(32) __bf16 Bf[12 * 512];

    int bi   = blockIdx.x;
    int b    = bi / N_;
    int lane = threadIdx.x & 31;
    int wave = threadIdx.x >> 5;
    int ntiles = 3 * nct;
    int cs8  = nct * 16;
    float px = p1[bi * 2 + 0], py = p1[bi * 2 + 1];

    v8f accr[5];
#pragma unroll
    for (int q = 0; q < 5; ++q) accr[q] = (v8f){0.f,0.f,0.f,0.f,0.f,0.f,0.f,0.f};

    int nch = Nsrc >> 5;
    for (int jc = 0; jc < nch; ++jc) {
        __syncthreads();                    // protect previous iteration reads
        // zero bin-weight tiles
#pragma unroll
        for (int q = 0; q < 3; ++q)
            ((unsigned int*)Aw)[threadIdx.x + q * 256] = 0u;
        // stage feature B-tiles
        int tot = nct * 512;
        for (int idx = threadIdx.x; idx < tot; idx += 256) {
            int ct = idx >> 9;
            int ln = (idx >> 4) & 31;
            int sl = idx & 15;
            int kk = (sl & 7) + ((sl & 8) ? 16 : 0) + ((ln & 16) ? 8 : 0);
            int cl = ct * 16 + (ln & 15);
            int j  = (jc << 5) + kk;
            Bf[idx] = feats[(long long)(b * Nsrc + j) * fstride + cl];
        }
        // bin weights for this 32-source chunk
        float w0 = 0.f, w1 = 0.f, w2v = 0.f, w3 = 0.f;
        int a0 = 0, a1 = 0, a2 = 0, a3 = 0; bool dup = true;
        if (threadIdx.x < 32) {
            int K = threadIdx.x;
            int gj = b * Nsrc + (jc << 5) + K;
            float rx = psrc[gj * 2 + 0] - px;
            float ry = psrc[gj * 2 + 1] - py;
            float d2 = rx * rx + ry * ry;
            bool safe = d2 > 1e-12f;
            float dist = safe ? sqrtf(d2) : 0.f;
            float th = atan2f(safe ? ry : 0.f, safe ? rx : 1.f);
            float valid = ((dist < 40.f) ? 1.f : 0.f) * msrc[gj];
            float rc = fminf(fmaxf(dist * (3.f / 40.f) - 0.5f, 0.f), 2.f);
            float r0f = floorf(rc);
            float fr = rc - r0f;
            int r0 = (int)r0f;
            int r1 = (r0 + 1 > 2) ? 2 : r0 + 1;
            float tc = th * (16.f / 6.28318530717958647692f) - 0.5f;
            tc -= floorf(tc * (1.f / 16.f)) * 16.f;
            float t0f = floorf(tc);
            float ft = tc - t0f;
            int t0 = ((int)t0f) & 15;
            int t1 = (t0 + 1) & 15;
            w0 = valid * (1.f - fr) * (1.f - ft);
            w1 = valid * (1.f - fr) * ft;
            w2v = valid * fr * (1.f - ft);
            w3 = valid * fr * ft;
            dup = (r1 == r0);
            if (dup) { w0 += w2v; w1 += w3; }
            int laneSel = (K & 8) ? 16 : 0;
            int slot = (K & 7) + ((K & 16) ? 8 : 0);
            int bin;
            bin = r0 * 16 + t0; a0 = ((bin >> 4) << 9) + ((bin & 15) + laneSel) * 16 + slot;
            bin = r0 * 16 + t1; a1 = ((bin >> 4) << 9) + ((bin & 15) + laneSel) * 16 + slot;
            bin = r1 * 16 + t0; a2 = ((bin >> 4) << 9) + ((bin & 15) + laneSel) * 16 + slot;
            bin = r1 * 16 + t1; a3 = ((bin >> 4) << 9) + ((bin & 15) + laneSel) * 16 + slot;
        }
        __syncthreads();                    // zeros + staging visible
        if (threadIdx.x < 32) {
            Aw[a0] = (__bf16)w0;
            Aw[a1] = (__bf16)w1;
            if (!dup) { Aw[a2] = (__bf16)w2v; Aw[a3] = (__bf16)w3; }
        }
        __syncthreads();                    // scatter visible
#pragma unroll
        for (int q = 0; q < 5; ++q) {
            int ti = wave + (q << 3);
            if (ti < ntiles) {
                int rt = ti / nct;
                int ct = ti - rt * nct;
                v16bf av = *(const v16bf*)&Aw[(rt << 9) + lane * 16];
                v16bf bv = *(const v16bf*)&Bf[(ct << 9) + lane * 16];
                accr[q] = __builtin_amdgcn_wmma_f32_16x16x32_bf16(
                    false, av, false, bv, (short)0, accr[q], false, false);
            }
        }
    }
    // store G in bf16, row-major [bin*cs8 + cs] (A-operand rows for k_gemm)
#pragma unroll
    for (int q = 0; q < 5; ++q) {
        int ti = wave + (q << 3);
        if (ti < ntiles) {
            int rt = ti / nct;
            int ct = ti - rt * nct;
            int col = (ct << 4) + (lane & 15);
            int rbase = (rt << 4) + ((lane & 16) ? 8 : 0);
            long long base = (long long)bi * astride;
#pragma unroll
            for (int v = 0; v < 8; ++v)
                Aout[base + (long long)(rbase + v) * cs8 + col] = (__bf16)accr[q][v];
        }
    }
}

// ---------------------------------------------------------------------------
// Dense GEMM: Y[row, coloff+col] = sum_k A[row,k]*B[k,col].  One wave per
// 16x16 output tile; A rows bf16 row-major, B pre-packed in operand layout.
// ---------------------------------------------------------------------------
__global__ __launch_bounds__(32)
void k_gemm(const __bf16* __restrict__ A, long long astride, int kchunks,
            const __bf16* __restrict__ Bm, int nctB,
            float* __restrict__ Y, int ystride, int ycoloff)
{
    int lane = threadIdx.x;
    int rt = blockIdx.x;
    int ct = blockIdx.y;
    v8f acc = (v8f){0.f,0.f,0.f,0.f,0.f,0.f,0.f,0.f};
    const __bf16* arow = A + (long long)(rt * 16 + (lane & 15)) * astride
                           + ((lane & 16) ? 16 : 0);
    const __bf16* bp = Bm + ((long long)ct << 9) + lane * 16;
    long long bstep = (long long)nctB << 9;
    for (int kc = 0; kc < kchunks; ++kc) {
        const __bf16* ap = arow + (kc << 5);
        union { v16bf v; v8bf h[2]; } au;
        au.h[0] = *(const v8bf*)(ap);
        au.h[1] = *(const v8bf*)(ap + 8);
        v16bf bv = *(const v16bf*)bp;
        bp += bstep;
        acc = __builtin_amdgcn_wmma_f32_16x16x32_bf16(
            false, au.v, false, bv, (short)0, acc, false, false);
    }
    int col = ycoloff + (ct << 4) + (lane & 15);
    int rbase = (lane & 16) ? 8 : 0;
#pragma unroll
    for (int v = 0; v < 8; ++v)
        Y[(rt * 16 + rbase + v) * ystride + col] = acc[v];
}

// ---------------------------------------------------------------------------
__global__ __launch_bounds__(256)
void k_combine(const float* __restrict__ Y, const float* __restrict__ bias,
               int biasMode, const float* __restrict__ prev,
               float* __restrict__ out, int ncols, int total)
{
    int idx = blockIdx.x * 256 + threadIdx.x;
    if (idx >= total) return;
    int col = idx % ncols;
    float v = Y[idx];
    if (biasMode == 0) { if (col >= 128) v += bias[(col - 128) >> 3]; }
    else               { v += bias[col >> 3]; }
    if (prev) v += prev[idx];
    out[idx] = v;
}

__global__ __launch_bounds__(256)
void k_relu_ext(const float* __restrict__ prev, __bf16* __restrict__ ext,
                long long astride, int cs8, int total)
{
    int idx = blockIdx.x * 256 + threadIdx.x;
    if (idx >= total) return;
    int bi = idx / cs8, c = idx - bi * cs8;
    float v = prev[idx];
    v = v > 0.f ? v : 0.f;
    ext[(long long)bi * astride + c] = (__bf16)v;
}

__global__ __launch_bounds__(256)
void k_finalize(const float* __restrict__ Y, const float* __restrict__ p1,
                const float* __restrict__ p0, float* __restrict__ outp)
{
    int bi = blockIdx.x * 256 + threadIdx.x;
    if (bi >= B_ * N_) return;
    float cx = Y[bi * 16 + 0] * (1.f / 128.f);
    float cy = Y[bi * 16 + 1] * (1.f / 128.f);
    float pcx = p1[bi * 2 + 0] + cx;
    float pcy = p1[bi * 2 + 1] + cy;
    outp[bi * 2 + 0] = pcx;
    outp[bi * 2 + 1] = pcy;
    outp[3072 + bi * 2 + 0] = (pcx - p0[bi * 2 + 0]) * (1.f / DT_);
    outp[3072 + bi * 2 + 1] = (pcy - p0[bi * 2 + 1]) * (1.f / DT_);
}

// ---------------------------------------------------------------------------
extern "C" void kernel_launch(void* const* d_in, const int* in_sizes, int n_in,
                              void* d_out, int out_size, void* d_ws, size_t ws_size,
                              hipStream_t stream)
{
    (void)in_sizes; (void)n_in; (void)out_size; (void)ws_size;
    const float* v0_enc        = (const float*)d_in[1];
    const float* p0            = (const float*)d_in[2];
    const float* v0            = (const float*)d_in[3];
    const float* a             = (const float*)d_in[4];
    const float* other         = (const float*)d_in[5];
    const float* box           = (const float*)d_in[6];
    const float* box_feats     = (const float*)d_in[7];
    const float* fmask         = (const float*)d_in[8];
    const float* bmask         = (const float*)d_in[9];
    const float* conv_fluid_W  = (const float*)d_in[10];
    const float* conv_obst_W   = (const float*)d_in[11];
    const float* dense_fluid_w1= (const float*)d_in[12];
    const float* dense_fluid_W2= (const float*)d_in[13];
    const float* dense_fluid_b2= (const float*)d_in[14];
    const float* conv1_W       = (const float*)d_in[15];
    const float* dense1_W      = (const float*)d_in[16];
    const float* dense1_b      = (const float*)d_in[17];
    const float* conv2_W       = (const float*)d_in[18];
    const float* dense2_W      = (const float*)d_in[19];
    const float* dense2_b      = (const float*)d_in[20];
    const float* conv3_W       = (const float*)d_in[21];
    const float* dense3_W      = (const float*)d_in[22];
    const float* dense3_b      = (const float*)d_in[23];
    const float* conv4_W       = (const float*)d_in[24];
    const float* dense4_W      = (const float*)d_in[25];
    const float* dense4_w2     = (const float*)d_in[27];

    char* ws = (char*)d_ws;
    float*  p1    = (float*)(ws + OFF_P1);
    __bf16* boxbf = (__bf16*)(ws + OFF_BOX);
    float*  Y     = (float*)(ws + OFF_Y);
    float*  out0  = (float*)(ws + OFF_OUT0);
    float*  out1  = (float*)(ws + OFF_OUT1);
    __bf16* Ao    = (__bf16*)(ws + OFF_AO);
    __bf16* Af    = (__bf16*)(ws + OFF_AF);
    __bf16* B0o   = (__bf16*)(ws + OFF_B0o);
    __bf16* B0f   = (__bf16*)(ws + OFF_B0f);
    __bf16* Bm1   = (__bf16*)(ws + OFF_B1);
    __bf16* Bm2   = (__bf16*)(ws + OFF_B2);
    __bf16* Bm3   = (__bf16*)(ws + OFF_B3);
    __bf16* Bm4   = (__bf16*)(ws + OFF_B4);
    float* outp = (float*)d_out;

    dim3 blk(256);
    // --- precompute kernel-tensor B operands ---
    k_prepB<<<(800*64+255)/256,   blk, 0, stream>>>(0,  800*64,  768,  16,  64,  1, conv_obst_W,  nullptr,        nullptr,        B0o);
    k_prepB<<<(2368*128+255)/256, blk, 0, stream>>>(1, 2368*128, 2304,  48, 128, 19, conv_fluid_W, dense_fluid_W2, dense_fluid_w1, B0f);
    k_prepB<<<(9408*128+255)/256, blk, 0, stream>>>(2, 9408*128, 9216, 192, 128, 24, conv1_W,      dense1_W,       nullptr,        Bm1);
    k_prepB<<<(6272*128+255)/256, blk, 0, stream>>>(3, 6272*128, 6144, 128, 128, 16, conv2_W,      dense2_W,       nullptr,        Bm2);
    k_prepB<<<(6272*128+255)/256, blk, 0, stream>>>(4, 6272*128, 6144, 128, 128, 16, conv3_W,      dense3_W,       nullptr,        Bm3);
    k_prepB<<<(6272*16+255)/256,  blk, 0, stream>>>(5, 6272*16,  6144, 128,  16, 16, conv4_W,      dense4_W,       dense4_w2,      Bm4);

    k_integrate<<<6, blk, 0, stream>>>(p0, v0, a, other, v0_enc, box_feats,
                                       p1, Af + 2304, Ao + 768, boxbf, outp + 6144);
    // --- layer 0: obstacle conv + fluid conv + dense (fused via col ranges) ---
    k_aggregate<<<B_*N_, blk, 0, stream>>>(p1, box, bmask, boxbf, 16, M_, 1, Ao, 800);
    k_gemm<<<dim3(96,4), dim3(32), 0, stream>>>(Ao, 800, 25, B0o, 4, Y, 192, 0);
    k_aggregate<<<B_*N_, blk, 0, stream>>>(p1, p1, fmask, Af + 2304, 2368, N_, 3, Af, 2368);
    k_gemm<<<dim3(96,8), dim3(32), 0, stream>>>(Af, 2368, 74, B0f, 8, Y, 192, 64);
    k_combine<<<1152, blk, 0, stream>>>(Y, dense_fluid_b2, 0, nullptr, out0, 192, 1536*192);
    // --- layer 1 (no residual: channel mismatch 24 -> 16) ---
    k_relu_ext<<<1152, blk, 0, stream>>>(out0, Af + 9216, 9408, 192, 1536*192);
    k_aggregate<<<B_*N_, blk, 0, stream>>>(p1, p1, fmask, Af + 9216, 9408, N_, 12, Af, 9408);
    k_gemm<<<dim3(96,8), dim3(32), 0, stream>>>(Af, 9408, 294, Bm1, 8, Y, 128, 0);
    k_combine<<<768, blk, 0, stream>>>(Y, dense1_b, 1, nullptr, out1, 128, 1536*128);
    // --- layer 2 (residual) ---
    k_relu_ext<<<768, blk, 0, stream>>>(out1, Af + 6144, 6272, 128, 1536*128);
    k_aggregate<<<B_*N_, blk, 0, stream>>>(p1, p1, fmask, Af + 6144, 6272, N_, 8, Af, 6272);
    k_gemm<<<dim3(96,8), dim3(32), 0, stream>>>(Af, 6272, 196, Bm2, 8, Y, 128, 0);
    k_combine<<<768, blk, 0, stream>>>(Y, dense2_b, 1, out1, out0, 128, 1536*128);
    // --- layer 3 (residual) ---
    k_relu_ext<<<768, blk, 0, stream>>>(out0, Af + 6144, 6272, 128, 1536*128);
    k_aggregate<<<B_*N_, blk, 0, stream>>>(p1, p1, fmask, Af + 6144, 6272, N_, 8, Af, 6272);
    k_gemm<<<dim3(96,8), dim3(32), 0, stream>>>(Af, 6272, 196, Bm3, 8, Y, 128, 0);
    k_combine<<<768, blk, 0, stream>>>(Y, dense3_b, 1, out0, out1, 128, 1536*128);
    // --- layer 4 (conv4 + dense4 -> position correction) ---
    k_relu_ext<<<768, blk, 0, stream>>>(out1, Af + 6144, 6272, 128, 1536*128);
    k_aggregate<<<B_*N_, blk, 0, stream>>>(p1, p1, fmask, Af + 6144, 6272, N_, 8, Af, 6272);
    k_gemm<<<dim3(96,1), dim3(32), 0, stream>>>(Af, 6272, 196, Bm4, 1, Y, 16, 0);
    k_finalize<<<6, blk, 0, stream>>>(Y, p1, p0, outp);
}
